// BitNetMLP_81046032875547
// MI455X (gfx1250) — matrix operand order repaired and checked by reading, not compiled, not run on hardware
//
#include <hip/hip_runtime.h>

typedef __attribute__((ext_vector_type(16))) __bf16 v16bf;
typedef __attribute__((ext_vector_type(8)))  __bf16 v8bf;
typedef __attribute__((ext_vector_type(4)))  __bf16 v4bf;
typedef __attribute__((ext_vector_type(8)))  float  v8f;
typedef __attribute__((ext_vector_type(4)))  int    v4i;

#define BM 64
#define BN 64
#define BK 32
#define LDT 40   // padded LDS row stride (elements): 80B -> conflict-free, 16B-aligned frags
#define TILE_ELEMS (BM * LDT)

// ---- CDNA5 async global->LDS copy (ASYNCcnt path), with safe fallback ------
#if defined(__has_builtin)
#  if __has_builtin(__builtin_amdgcn_global_load_async_to_lds_b128) && \
      __has_builtin(__builtin_amdgcn_s_wait_asynccnt)
#    define BN_HAVE_ASYNC 1
#  endif
#endif
#ifndef BN_HAVE_ASYNC
#  define BN_HAVE_ASYNC 0
#endif

__device__ __forceinline__ void bn_cp16(const __bf16* g, __bf16* l) {
#if BN_HAVE_ASYNC
  __builtin_amdgcn_global_load_async_to_lds_b128(
      (__attribute__((address_space(1))) v4i*)g,
      (__attribute__((address_space(3))) v4i*)l, 0, 0);
#else
  *(v8bf*)l = *(const v8bf*)g;
#endif
}
__device__ __forceinline__ void bn_cp_wait() {
#if BN_HAVE_ASYNC
  __builtin_amdgcn_s_wait_asynccnt(0);
#endif
}

// ---------------------------------------------------------------------------
// Deterministic |w| reduction: per-block partials (fixed grid), then tree sum.
// ---------------------------------------------------------------------------
__global__ __launch_bounds__(256) void bn_abssum_partial(
    const float* __restrict__ w, long long n4, float* __restrict__ partial) {
  __shared__ float red[256];
  const int tid = threadIdx.x;
  long long i = ((long long)blockIdx.x * 256 + tid);
  const long long stride = (long long)gridDim.x * 256;
  float s = 0.0f;
  for (; i < n4; i += stride) {
    float4 v = ((const float4*)w)[i];
    s += fabsf(v.x) + fabsf(v.y) + fabsf(v.z) + fabsf(v.w);
  }
  red[tid] = s;
  __syncthreads();
  for (int off = 128; off > 0; off >>= 1) {
    if (tid < off) red[tid] += red[tid + off];
    __syncthreads();
  }
  if (tid == 0) partial[blockIdx.x] = red[0];
}

__global__ __launch_bounds__(256) void bn_finalize_thr(
    const float* __restrict__ partial, float factor, float* __restrict__ thr_out) {
  __shared__ float red[256];
  const int tid = threadIdx.x;
  float s = 0.0f;
  for (int i = tid; i < 1024; i += 256) s += partial[i];
  red[tid] = s;
  __syncthreads();
  for (int off = 128; off > 0; off >>= 1) {
    if (tid < off) red[tid] += red[tid + off];
    __syncthreads();
  }
  if (tid == 0) thr_out[0] = red[0] * factor;  // factor = 0.7 / numel
}

// Ternary quantize fp32 -> bf16 {-1,0,+1}
__global__ __launch_bounds__(256) void bn_quantize(
    const float* __restrict__ w, const float* __restrict__ thrp,
    __bf16* __restrict__ q, long long n) {
  const float thr = thrp[0];
  const long long stride = (long long)gridDim.x * 256 * 4;
  for (long long i = ((long long)blockIdx.x * 256 + threadIdx.x) * 4; i < n; i += stride) {
    float4 v = *(const float4*)(w + i);
    v4bf o;
    o[0] = (__bf16)((fabsf(v.x) > thr) ? copysignf(1.0f, v.x) : 0.0f);
    o[1] = (__bf16)((fabsf(v.y) > thr) ? copysignf(1.0f, v.y) : 0.0f);
    o[2] = (__bf16)((fabsf(v.z) > thr) ? copysignf(1.0f, v.z) : 0.0f);
    o[3] = (__bf16)((fabsf(v.w) > thr) ? copysignf(1.0f, v.w) : 0.0f);
    *(v4bf*)(q + i) = o;
  }
}

__global__ __launch_bounds__(256) void bn_f32_to_bf16(
    const float* __restrict__ x, __bf16* __restrict__ y, long long n) {
  const long long stride = (long long)gridDim.x * 256 * 4;
  for (long long i = ((long long)blockIdx.x * 256 + threadIdx.x) * 4; i < n; i += stride) {
    float4 v = *(const float4*)(x + i);
    v4bf o;
    o[0] = (__bf16)v.x; o[1] = (__bf16)v.y; o[2] = (__bf16)v.z; o[3] = (__bf16)v.w;
    *(v4bf*)(y + i) = o;
  }
}

// ---------------------------------------------------------------------------
// Fragment loaders (bf16 16x16x32 WMMA, wave32, per ISA layout tables)
// ---------------------------------------------------------------------------
__device__ __forceinline__ v16bf bn_frag_a(const __bf16* row, int kh) {
  // lanes 0-15: K {0..7, 16..23}; lanes 16-31: K {8..15, 24..31}
  const __bf16* p = row + kh * 8;
  v8bf lo = *(const v8bf*)(p);
  v8bf hi = *(const v8bf*)(p + 16);
  return __builtin_shufflevector(lo, hi, 0,1,2,3,4,5,6,7,8,9,10,11,12,13,14,15);
}
__device__ __forceinline__ v16bf bn_frag_b(const __bf16* row, int kh) {
  // lanes 0-15: K {0..7, 16..23}; lanes 16-31: K {8..15, 24..31}
  const __bf16* p = row + kh * 8;
  v8bf lo = *(const v8bf*)(p);
  v8bf hi = *(const v8bf*)(p + 16);
  return __builtin_shufflevector(lo, hi, 0,1,2,3,4,5,6,7,8,9,10,11,12,13,14,15);
}

// ---------------------------------------------------------------------------
// Fused gate+up GEMM: H[m,n] = silu(gs[n]*X.Wg^T) * (us[n]*X.Wu^T), bf16 out
// 256 thr (8 waves), tile 64M x 64N, BK=32, double-buffered LDS, async copies.
// Wave w: mtile = (w&3)*16, ntiles = (w>>2)*32 + {0,16}; 4 WMMA / k-step.
// ---------------------------------------------------------------------------
__global__ __launch_bounds__(256) void bn_gateup_gemm(
    const __bf16* __restrict__ X,  const __bf16* __restrict__ Wg,
    const __bf16* __restrict__ Wu, const float* __restrict__ gsc,
    const float* __restrict__ usc, __bf16* __restrict__ H,
    int K, int N) {
  __shared__ __bf16 sA[2 * TILE_ELEMS];
  __shared__ __bf16 sG[2 * TILE_ELEMS];
  __shared__ __bf16 sU[2 * TILE_ELEMS];

  const int tid  = threadIdx.x;
  const int wave = tid >> 5;
  const int lane = tid & 31;
  const int r    = lane & 15;
  const int kh   = lane >> 4;
  const int mt   = (wave & 3) << 4;        // 0,16,32,48
  const int np   = (wave >> 2) << 5;       // 0,32
  const long long Mb = (long long)blockIdx.y * BM;
  const long long Nb = (long long)blockIdx.x * BN;

  v8f ag0 = {0,0,0,0,0,0,0,0}, ag1 = {0,0,0,0,0,0,0,0};
  v8f au0 = {0,0,0,0,0,0,0,0}, au1 = {0,0,0,0,0,0,0,0};

  const int row   = tid >> 2;              // 0..63
  const int chunk = (tid & 3) << 3;        // 0,8,16,24 (elements)
  const int st    = row * LDT + chunk;     // staging offset in a tile
  const __bf16* gx = X  + (Mb + row) * (long long)K + chunk;
  const __bf16* gg = Wg + (Nb + row) * (long long)K + chunk;
  const __bf16* gu = Wu + (Nb + row) * (long long)K + chunk;

  const int fAo  = (mt + r) * LDT;         // fragment row offsets in a tile
  const int fB0o = (np + r) * LDT;
  const int fB1o = (np + 16 + r) * LDT;

  const int nk = K / BK;

  // Prologue: stage tile 0 into buffer 0.
  bn_cp16(gx, &sA[st]);
  bn_cp16(gg, &sG[st]);
  bn_cp16(gu, &sU[st]);
  gx += BK; gg += BK; gu += BK;
  bn_cp_wait();
  __syncthreads();

  for (int k = 0; k < nk; ++k) {
    const int cur = (k & 1) * TILE_ELEMS;
    const int nxt = ((k + 1) & 1) * TILE_ELEMS;

    if (k + 1 < nk) {  // stage next tile into the other buffer (overlaps WMMA)
      bn_cp16(gx, &sA[nxt + st]);
      bn_cp16(gg, &sG[nxt + st]);
      bn_cp16(gu, &sU[nxt + st]);
      gx += BK; gg += BK; gu += BK;
    }

    v16bf a  = bn_frag_a(&sA[cur + fAo],  kh);
    v16bf g0 = bn_frag_b(&sG[cur + fB0o], kh);
    v16bf g1 = bn_frag_b(&sG[cur + fB1o], kh);
    v16bf u0 = bn_frag_b(&sU[cur + fB0o], kh);
    v16bf u1 = bn_frag_b(&sU[cur + fB1o], kh);
    ag0 = __builtin_amdgcn_wmma_f32_16x16x32_bf16(false, a, false, g0, (short)0, ag0, false, false);
    ag1 = __builtin_amdgcn_wmma_f32_16x16x32_bf16(false, a, false, g1, (short)0, ag1, false, false);
    au0 = __builtin_amdgcn_wmma_f32_16x16x32_bf16(false, a, false, u0, (short)0, au0, false, false);
    au1 = __builtin_amdgcn_wmma_f32_16x16x32_bf16(false, a, false, u1, (short)0, au1, false, false);

    if (k + 1 < nk) bn_cp_wait();
    __syncthreads();
  }

  // Epilogue: per-column scales, SiLU(gate)*up, store bf16 hidden.
  for (int t = 0; t < 2; ++t) {
    v8f vg = t ? ag1 : ag0;
    v8f vu = t ? au1 : au0;
    const long long n = Nb + np + t * 16 + r;   // C layout: N = lane%16
    const float gsv = gsc[n];
    const float usv = usc[n];
    #pragma unroll
    for (int i = 0; i < 8; ++i) {
      const long long m = Mb + mt + i + 8 * kh; // C layout: M = i + 8*(lane/16)
      const float gv = vg[i] * gsv;
      const float hv = (gv / (1.0f + __expf(-gv))) * (vu[i] * usv);
      H[m * (long long)N + n] = (__bf16)hv;
    }
  }
}

// ---------------------------------------------------------------------------
// Down GEMM: Out[m,n] = dsc[n] * (H . Wd^T), fp32 out. Same pipeline.
// ---------------------------------------------------------------------------
__global__ __launch_bounds__(256) void bn_down_gemm(
    const __bf16* __restrict__ Hb, const __bf16* __restrict__ Wd,
    const float* __restrict__ dscale, float* __restrict__ Out,
    int K, int N) {
  __shared__ __bf16 sA[2 * TILE_ELEMS];
  __shared__ __bf16 sB[2 * TILE_ELEMS];

  const int tid  = threadIdx.x;
  const int wave = tid >> 5;
  const int lane = tid & 31;
  const int r    = lane & 15;
  const int kh   = lane >> 4;
  const int mt   = (wave & 3) << 4;
  const int np   = (wave >> 2) << 5;
  const long long Mb = (long long)blockIdx.y * BM;
  const long long Nb = (long long)blockIdx.x * BN;

  v8f a0 = {0,0,0,0,0,0,0,0}, a1 = {0,0,0,0,0,0,0,0};

  const int row   = tid >> 2;
  const int chunk = (tid & 3) << 3;
  const int st    = row * LDT + chunk;
  const __bf16* gh = Hb + (Mb + row) * (long long)K + chunk;
  const __bf16* gw = Wd + (Nb + row) * (long long)K + chunk;

  const int fAo  = (mt + r) * LDT;
  const int fB0o = (np + r) * LDT;
  const int fB1o = (np + 16 + r) * LDT;

  const int nk = K / BK;

  bn_cp16(gh, &sA[st]);
  bn_cp16(gw, &sB[st]);
  gh += BK; gw += BK;
  bn_cp_wait();
  __syncthreads();

  for (int k = 0; k < nk; ++k) {
    const int cur = (k & 1) * TILE_ELEMS;
    const int nxt = ((k + 1) & 1) * TILE_ELEMS;

    if (k + 1 < nk) {
      bn_cp16(gh, &sA[nxt + st]);
      bn_cp16(gw, &sB[nxt + st]);
      gh += BK; gw += BK;
    }

    v16bf a  = bn_frag_a(&sA[cur + fAo],  kh);
    v16bf b0 = bn_frag_b(&sB[cur + fB0o], kh);
    v16bf b1 = bn_frag_b(&sB[cur + fB1o], kh);
    a0 = __builtin_amdgcn_wmma_f32_16x16x32_bf16(false, a, false, b0, (short)0, a0, false, false);
    a1 = __builtin_amdgcn_wmma_f32_16x16x32_bf16(false, a, false, b1, (short)0, a1, false, false);

    if (k + 1 < nk) bn_cp_wait();
    __syncthreads();
  }

  for (int t = 0; t < 2; ++t) {
    v8f v = t ? a1 : a0;
    const long long n = Nb + np + t * 16 + r;
    const float dsv = dscale[n];
    #pragma unroll
    for (int i = 0; i < 8; ++i) {
      const long long m = Mb + mt + i + 8 * kh;
      Out[m * (long long)N + n] = v[i] * dsv;
    }
  }
}

// ---------------------------------------------------------------------------
extern "C" void kernel_launch(void* const* d_in, const int* in_sizes, int n_in,
                              void* d_out, int out_size, void* d_ws, size_t ws_size,
                              hipStream_t stream) {
  (void)n_in; (void)out_size; (void)ws_size;
  const float* x   = (const float*)d_in[0];
  const float* gw  = (const float*)d_in[1];
  const float* gsc = (const float*)d_in[2];
  const float* uw  = (const float*)d_in[3];
  const float* usc = (const float*)d_in[4];
  const float* dw  = (const float*)d_in[5];
  const float* dsc = (const float*)d_in[6];

  const long long hidden = in_sizes[2];                 // 16384
  const long long dim    = in_sizes[6];                 // 4096
  const long long M      = (long long)in_sizes[0] / dim;// 4096
  const long long wn     = hidden * dim;                // weight numel

  // Workspace layout
  char* ws = (char*)d_ws;
  float* thr     = (float*)ws;                          // 3 thresholds
  float* partial = (float*)(ws + 256);                  // 3 * 1024 partials
  size_t off = 256 + 3 * 1024 * sizeof(float);
  off = (off + 255) & ~(size_t)255;
  __bf16* xq = (__bf16*)(ws + off); off += (size_t)(M * dim) * 2;      off = (off + 255) & ~(size_t)255;
  __bf16* gq = (__bf16*)(ws + off); off += (size_t)wn * 2;             off = (off + 255) & ~(size_t)255;
  __bf16* uq = (__bf16*)(ws + off); off += (size_t)wn * 2;             off = (off + 255) & ~(size_t)255;
  __bf16* dq = (__bf16*)(ws + off); off += (size_t)wn * 2;             off = (off + 255) & ~(size_t)255;
  __bf16* hid = (__bf16*)(ws + off);

  // 1) deterministic abs-mean thresholds
  bn_abssum_partial<<<1024, 256, 0, stream>>>(gw, wn / 4, partial);
  bn_abssum_partial<<<1024, 256, 0, stream>>>(uw, wn / 4, partial + 1024);
  bn_abssum_partial<<<1024, 256, 0, stream>>>(dw, wn / 4, partial + 2048);
  const float factor = 0.7f / (float)wn;
  bn_finalize_thr<<<1, 256, 0, stream>>>(partial,        factor, thr);
  bn_finalize_thr<<<1, 256, 0, stream>>>(partial + 1024, factor, thr + 1);
  bn_finalize_thr<<<1, 256, 0, stream>>>(partial + 2048, factor, thr + 2);

  // 2) ternary-quantize weights to bf16; convert x to bf16
  bn_quantize<<<2048, 256, 0, stream>>>(gw, thr,     gq, wn);
  bn_quantize<<<2048, 256, 0, stream>>>(uw, thr + 1, uq, wn);
  bn_quantize<<<2048, 256, 0, stream>>>(dw, thr + 2, dq, wn);
  bn_f32_to_bf16<<<1024, 256, 0, stream>>>(x, xq, M * dim);

  // 3) fused gate+up WMMA GEMM -> bf16 hidden
  dim3 g1((unsigned)(hidden / BN), (unsigned)(M / BM));
  bn_gateup_gemm<<<g1, 256, 0, stream>>>(xq, gq, uq, gsc, usc, hid, (int)dim, (int)hidden);

  // 4) down WMMA GEMM -> fp32 output
  dim3 g2((unsigned)(dim / BN), (unsigned)(M / BM));
  bn_down_gemm<<<g2, 256, 0, stream>>>(hid, dq, dsc, (float*)d_out, (int)hidden, (int)dim);
}
